// Mamba2Block_29875792511252
// MI455X (gfx1250) — matrix operand order repaired and checked by reading, not compiled, not run on hardware
//
#include <hip/hip_runtime.h>
#include <hip/hip_bf16.h>
#include <stdint.h>

typedef __attribute__((ext_vector_type(16))) __bf16 v16bf;
typedef __attribute__((ext_vector_type(8)))  __bf16 v8bf;
typedef __attribute__((ext_vector_type(8)))  float  v8f;
typedef __attribute__((ext_vector_type(4)))  int    v4i;

#define DIMX   1024
#define STATEX 64
#define KCONV  4
#define INNERX 2048
#define BBX    2
#define LLX    1024
#define NROWS  (BBX*LLX)          /* 2048 GEMM rows (B*L) */
#define EPSX   1.1920929e-07f

// ---------------------------------------------------------------- async copy helpers
#if defined(__gfx1250__) && __has_builtin(__builtin_amdgcn_global_load_async_to_lds_b128)
#define ASYNC_COPY 1
#else
#define ASYNC_COPY 0
#endif

__device__ __forceinline__ void copy_b128_to_lds(const void* g, void* l) {
#if ASYNC_COPY
  __builtin_amdgcn_global_load_async_to_lds_b128(
      (__attribute__((address_space(1))) v4i*)(uintptr_t)g,
      (__attribute__((address_space(3))) v4i*)(uint32_t)(uintptr_t)l,
      0, 0);
#else
  *(v8bf*)l = *(const v8bf*)g;
#endif
}

__device__ __forceinline__ void async_wait_le3() {
#if ASYNC_COPY
#if __has_builtin(__builtin_amdgcn_s_wait_asynccnt)
  __builtin_amdgcn_s_wait_asynccnt(3);
#else
  asm volatile("s_wait_asynccnt 0x3" ::: "memory");
#endif
#endif
}

__device__ __forceinline__ void async_wait_le0() {
#if ASYNC_COPY
#if __has_builtin(__builtin_amdgcn_s_wait_asynccnt)
  __builtin_amdgcn_s_wait_asynccnt(0);
#else
  asm volatile("s_wait_asynccnt 0x0" ::: "memory");
#endif
#endif
}

// ---------------------------------------------------------------- weight convert + transpose
// in: [Kdim][Ndim] f32 (row-major)  ->  out: [Ndim][Kdim] bf16 (row-major)
__global__ void cvt_f32_bf16_T(const float* __restrict__ in, __bf16* __restrict__ out,
                               int Kdim, int Ndim) {
  int i = blockIdx.x * 256 + threadIdx.x;
  if (i >= Kdim * Ndim) return;
  int k = i % Kdim;
  int n = i / Kdim;
  out[i] = (__bf16)in[k * Ndim + n];
}

// ---------------------------------------------------------------- RMSNorm -> bf16
__global__ __launch_bounds__(256)
void rmsnorm_kernel(const float* __restrict__ x, const float* __restrict__ w,
                    __bf16* __restrict__ xn) {
  const int row = blockIdx.x;
  const float* xr = x + row * DIMX;
  float ss = 0.f;
  for (int i = threadIdx.x; i < DIMX; i += 256) { float v = xr[i]; ss += v * v; }
  __shared__ float red[256];
  red[threadIdx.x] = ss;
  __syncthreads();
  for (int s = 128; s > 0; s >>= 1) {
    if (threadIdx.x < s) red[threadIdx.x] += red[threadIdx.x + s];
    __syncthreads();
  }
  const float scale = rsqrtf(red[0] / (float)DIMX + EPSX);
  for (int i = threadIdx.x; i < DIMX; i += 256)
    xn[row * DIMX + i] = (__bf16)(xr[i] * scale * w[i]);
}

// ---------------------------------------------------------------- bf16 WMMA GEMM
// C[M,N] = A[M,K](bf16,row-major) * Bt[N,K](bf16, pre-transposed weight), fp32 accum.
// Block tile 128x64, 8 waves (4x2), each wave a 32x32 output (4 WMMA accumulators).
// mode 0: store C ; mode 1: softplus(C + bias[col]) ; mode 2: C + resid[r,c]
__global__ __launch_bounds__(256)
void gemm_bf16_wmma(const __bf16* __restrict__ Ag, const __bf16* __restrict__ Bt,
                    float* __restrict__ Cg, int M, int N, int K,
                    int mode, const float* __restrict__ bias,
                    const float* __restrict__ resid) {
  __shared__ __align__(64) __bf16 As[2][128][32];  // [buf][m][k]  8KB/buf
  __shared__ __align__(64) __bf16 Bs[2][64][32];   // [buf][n][k]  4KB/buf
  const int tid  = threadIdx.x;
  const int wv   = tid >> 5;
  const int lane = tid & 31;
  const int row0 = blockIdx.y << 7;       // 128-row tiles
  const int col0 = blockIdx.x << 6;       // 64-col tiles
  const int moff = (wv & 3) << 5;         // wave M offset: 0,32,64,96
  const int noff = (wv >> 2) << 5;        // wave N offset: 0,32

  // staging: thread -> three 16B chunks (A rows lm and lm+64, B row lm)
  const int lm = tid >> 2;                // 0..63
  const int lk = (tid & 3) << 3;          // 0,8,16,24
  const __bf16* aSrc0 = Ag + (row0 + lm) * K + lk;
  const __bf16* aSrc1 = Ag + (row0 + lm + 64) * K + lk;
  const __bf16* bSrc  = Bt + (col0 + lm) * K + lk;

  v8f acc00 = {}, acc01 = {}, acc10 = {}, acc11 = {};
  const int Mr0  = moff + (lane & 15);          // A frag rows (two subtiles)
  const int Mr1  = Mr0 + 16;
  const int klo  = (lane < 16) ? 0 : 16;        // contiguous A-layout half-wave K base
  const int kb   = (lane < 16) ? 0 : 16;        // contiguous B-layout half-wave K base
  const int Nr0  = noff + (lane & 15);          // B frag rows (two subtiles)
  const int Nr1  = Nr0 + 16;

  const int S = K >> 5;                   // K-steps of 32
  // prologue: stage 0
  copy_b128_to_lds(aSrc0, &As[0][lm][lk]);
  copy_b128_to_lds(aSrc1, &As[0][lm + 64][lk]);
  copy_b128_to_lds(bSrc,  &Bs[0][lm][lk]);

  for (int s = 0; s < S; ++s) {
    if (s + 1 < S) {
      const int koff = (s + 1) << 5;
      const int nb = (s + 1) & 1;
      copy_b128_to_lds(aSrc0 + koff, &As[nb][lm][lk]);
      copy_b128_to_lds(aSrc1 + koff, &As[nb][lm + 64][lk]);
      copy_b128_to_lds(bSrc + koff,  &Bs[nb][lm][lk]);
      async_wait_le3();                   // stage s complete; s+1 still in flight
    } else {
      async_wait_le0();
    }
    __syncthreads();

    const __bf16 (*A_)[32] = As[s & 1];
    const __bf16 (*B_)[32] = Bs[s & 1];
    v8bf a0lo = *(const v8bf*)(&A_[Mr0][klo]);
    v8bf a0hi = *(const v8bf*)(&A_[Mr0][klo + 8]);
    v8bf a1lo = *(const v8bf*)(&A_[Mr1][klo]);
    v8bf a1hi = *(const v8bf*)(&A_[Mr1][klo + 8]);
    v16bf a0 = __builtin_shufflevector(a0lo, a0hi, 0,1,2,3,4,5,6,7,8,9,10,11,12,13,14,15);
    v16bf a1 = __builtin_shufflevector(a1lo, a1hi, 0,1,2,3,4,5,6,7,8,9,10,11,12,13,14,15);
    v16bf b0 = *(const v16bf*)(&B_[Nr0][kb]);
    v16bf b1 = *(const v16bf*)(&B_[Nr1][kb]);
    acc00 = __builtin_amdgcn_wmma_f32_16x16x32_bf16(false, a0, false, b0, (short)0, acc00, false, false);
    acc01 = __builtin_amdgcn_wmma_f32_16x16x32_bf16(false, a0, false, b1, (short)0, acc01, false, false);
    acc10 = __builtin_amdgcn_wmma_f32_16x16x32_bf16(false, a1, false, b0, (short)0, acc10, false, false);
    acc11 = __builtin_amdgcn_wmma_f32_16x16x32_bf16(false, a1, false, b1, (short)0, acc11, false, false);
    __syncthreads();                      // protect buffer before it is refilled
  }

  // Epilogue: D layout -> M = (lane<16?0:8)+v, N = lane&15
  const int rb0 = row0 + moff + ((lane < 16) ? 0 : 8);
  const int rb1 = rb0 + 16;
  const int c0  = col0 + noff + (lane & 15);
  const int c1  = c0 + 16;
  float bias0 = 0.f, bias1 = 0.f;
  if (mode == 1) { bias0 = bias[c0]; bias1 = bias[c1]; }
#pragma unroll
  for (int v = 0; v < 8; ++v) {
    float v00 = acc00[v], v01 = acc01[v], v10 = acc10[v], v11 = acc11[v];
    int r0 = rb0 + v, r1 = rb1 + v;
    if (mode == 1) {
      float t;
      t = v00 + bias0; v00 = (t > 20.f) ? t : log1pf(__expf(t));
      t = v01 + bias1; v01 = (t > 20.f) ? t : log1pf(__expf(t));
      t = v10 + bias0; v10 = (t > 20.f) ? t : log1pf(__expf(t));
      t = v11 + bias1; v11 = (t > 20.f) ? t : log1pf(__expf(t));
    } else if (mode == 2) {
      v00 += resid[r0 * N + c0];
      v01 += resid[r0 * N + c1];
      v10 += resid[r1 * N + c0];
      v11 += resid[r1 * N + c1];
    }
    Cg[r0 * N + c0] = v00;
    Cg[r0 * N + c1] = v01;
    Cg[r1 * N + c0] = v10;
    Cg[r1 * N + c1] = v11;
  }
}

// ---------------------------------------------------------------- causal dwconv + SiLU
__global__ __launch_bounds__(256)
void conv_silu_kernel(const float* __restrict__ xz, const float* __restrict__ cw,
                      const float* __restrict__ cb, __bf16* __restrict__ ubf) {
  int idx = blockIdx.x * 256 + threadIdx.x;           // over B*L*INNER = 2^22
  int c = idx & (INNERX - 1);
  int t = (idx >> 11) & (LLX - 1);
  int b = idx >> 21;
  float acc = cb[c];
#pragma unroll
  for (int k = 0; k < KCONV; ++k) {
    int tt = t + k - (KCONV - 1);
    if (tt >= 0) acc += cw[c * KCONV + k] * xz[((b * LLX + tt) << 12) + c];
  }
  float s = acc * (1.f / (1.f + __expf(-acc)));
  ubf[idx] = (__bf16)s;
}

// ---------------------------------------------------------------- selective scan + gate
// block = (64 states, 4 channels); one fp32 recurrence per (b,d,n) thread.
__global__ __launch_bounds__(256)
void scan_gate_kernel(const float* __restrict__ dt, const __bf16* __restrict__ ubf,
                      const float* __restrict__ Bx, const float* __restrict__ Cx,
                      const float* __restrict__ xz, const float* __restrict__ A_log,
                      const float* __restrict__ Dp, __bf16* __restrict__ yg) {
  const int n  = threadIdx.x;          // 0..63  (STATE)
  const int dj = threadIdx.y;          // 0..3
  const int d  = blockIdx.x * 4 + dj;
  const int b  = blockIdx.y;
  const float Aval = -__expf(A_log[d * STATEX + n]);
  const float Dval = Dp[d];
  __shared__ float red[4][2];
  float h = 0.f;
  for (int t = 0; t < LLX; ++t) {
    const int base = b * LLX + t;
    float dtv = dt[base * INNERX + d];
    float uv  = (float)ubf[base * INNERX + d];
    float bx  = Bx[base * STATEX + n];
    float cx  = Cx[base * STATEX + n];
    float dA  = __expf(dtv * Aval);
    h = dA * h + (dtv * uv) * bx;
    float p = h * cx;
#pragma unroll
    for (int off = 16; off > 0; off >>= 1) p += __shfl_down(p, off, 32);
    if ((n & 31) == 0) red[dj][n >> 5] = p;
    __syncthreads();
    if (n == 0) {
      float y = red[dj][0] + red[dj][1] + Dval * uv;
      float z = xz[(base << 12) + INNERX + d];
      float g = z * (1.f / (1.f + __expf(-z)));
      yg[base * INNERX + d] = (__bf16)(y * g);
    }
    __syncthreads();
  }
}

// ---------------------------------------------------------------- launcher
extern "C" void kernel_launch(void* const* d_in, const int* in_sizes, int n_in,
                              void* d_out, int out_size, void* d_ws, size_t ws_size,
                              hipStream_t stream) {
  (void)in_sizes; (void)n_in; (void)out_size; (void)ws_size;
  const float* x     = (const float*)d_in[0];
  const float* normw = (const float*)d_in[1];
  const float* W_in  = (const float*)d_in[2];
  const float* convw = (const float*)d_in[3];
  const float* convb = (const float*)d_in[4];
  const float* W_dt  = (const float*)d_in[5];
  const float* b_dt  = (const float*)d_in[6];
  const float* A_log = (const float*)d_in[7];
  const float* W_B   = (const float*)d_in[8];
  const float* W_C   = (const float*)d_in[9];
  const float* Dp    = (const float*)d_in[10];
  const float* W_out = (const float*)d_in[11];
  float* out = (float*)d_out;

  char* ws = (char*)d_ws;
  size_t off = 0;
  auto alloc = [&](size_t bytes) -> void* {
    void* p = ws + off;
    off += (bytes + 255) & ~(size_t)255;
    return p;
  };
  // weights stored TRANSPOSED ([N][K]) in bf16
  __bf16* W_in_t  = (__bf16*)alloc((size_t)DIMX * 2 * INNERX * 2);
  __bf16* W_dt_t  = (__bf16*)alloc((size_t)INNERX * INNERX * 2);
  __bf16* W_B_t   = (__bf16*)alloc((size_t)INNERX * STATEX * 2);
  __bf16* W_C_t   = (__bf16*)alloc((size_t)INNERX * STATEX * 2);
  __bf16* W_out_t = (__bf16*)alloc((size_t)INNERX * DIMX * 2);
  __bf16* xn_b    = (__bf16*)alloc((size_t)NROWS * DIMX * 2);
  float*  xz_f    = (float*) alloc((size_t)NROWS * 2 * INNERX * 4);
  __bf16* u_b     = (__bf16*)alloc((size_t)NROWS * INNERX * 2);
  float*  dt_f    = (float*) alloc((size_t)NROWS * INNERX * 4);
  float*  Bx_f    = (float*) alloc((size_t)NROWS * STATEX * 4);
  float*  Cx_f    = (float*) alloc((size_t)NROWS * STATEX * 4);
  __bf16* yg_b    = (__bf16*)alloc((size_t)NROWS * INNERX * 2);

  // 1) weights -> bf16, transposed to [N][K]
  {
    int n;
    n = DIMX * 2 * INNERX;  cvt_f32_bf16_T<<<(n + 255) / 256, 256, 0, stream>>>(W_in,  W_in_t,  DIMX,   2 * INNERX);
    n = INNERX * INNERX;    cvt_f32_bf16_T<<<(n + 255) / 256, 256, 0, stream>>>(W_dt,  W_dt_t,  INNERX, INNERX);
    n = INNERX * STATEX;    cvt_f32_bf16_T<<<(n + 255) / 256, 256, 0, stream>>>(W_B,   W_B_t,   INNERX, STATEX);
    n = INNERX * STATEX;    cvt_f32_bf16_T<<<(n + 255) / 256, 256, 0, stream>>>(W_C,   W_C_t,   INNERX, STATEX);
    n = INNERX * DIMX;      cvt_f32_bf16_T<<<(n + 255) / 256, 256, 0, stream>>>(W_out, W_out_t, INNERX, DIMX);
  }
  // 2) RMSNorm
  rmsnorm_kernel<<<NROWS, 256, 0, stream>>>(x, normw, xn_b);
  // 3) xz = xn @ W_in   [2048 x 4096], K=1024
  gemm_bf16_wmma<<<dim3(2 * INNERX / 64, NROWS / 128), 256, 0, stream>>>(
      xn_b, W_in_t, xz_f, NROWS, 2 * INNERX, DIMX, 0, nullptr, nullptr);
  // 4) u = silu(causal dwconv(xz[:, :INNER]))
  conv_silu_kernel<<<(BBX * LLX * INNERX) / 256, 256, 0, stream>>>(xz_f, convw, convb, u_b);
  // 5) dt = softplus(u @ W_dt + b_dt)   [2048 x 2048], K=2048
  gemm_bf16_wmma<<<dim3(INNERX / 64, NROWS / 128), 256, 0, stream>>>(
      u_b, W_dt_t, dt_f, NROWS, INNERX, INNERX, 1, b_dt, nullptr);
  // 6) Bx = u @ W_B ; Cx = u @ W_C   [2048 x 64], K=2048
  gemm_bf16_wmma<<<dim3(1, NROWS / 128), 256, 0, stream>>>(
      u_b, W_B_t, Bx_f, NROWS, STATEX, INNERX, 0, nullptr, nullptr);
  gemm_bf16_wmma<<<dim3(1, NROWS / 128), 256, 0, stream>>>(
      u_b, W_C_t, Cx_f, NROWS, STATEX, INNERX, 0, nullptr, nullptr);
  // 7) selective scan + silu(z) gate -> bf16 y
  scan_gate_kernel<<<dim3(INNERX / 4, BBX), dim3(64, 4), 0, stream>>>(
      dt_f, u_b, Bx_f, Cx_f, xz_f, A_log, Dp, yg_b);
  // 8) out = y @ W_out + x   [2048 x 1024], K=2048
  gemm_bf16_wmma<<<dim3(DIMX / 64, NROWS / 128), 256, 0, stream>>>(
      yg_b, W_out_t, out, NROWS, DIMX, INNERX, 2, nullptr, x);
}
